// GaussJordanGF2_42941083025869
// MI455X (gfx1250) — compile-verified, matrix operand order checked by reading, not run on hardware
//
#include <hip/hip_runtime.h>

// Gauss-Jordan over GF(2), bit-packed in LDS, single workgroup (one WGP).
// N=1024 columns, R=512 rows, K=512. Rows packed as 32 x u32 words (64KB LDS).
// Phase 1: 1024 sequential column-elimination steps (ds atomics + XOR).
// Phase 2: colsum = ones^T * H via v_wmma_f32_16x16x32_f16, then 512 column-swap steps.
// Phase 3: assemble G_nsys = scatter([I | P^T], col_swap) + src_index.

typedef _Float16 h16 __attribute__((ext_vector_type(16)));
typedef float    f8  __attribute__((ext_vector_type(8)));

#define NROW 512
#define NCOL 1024
#define NWRD 32   // u32 words per packed row

__global__ __launch_bounds__(512)
void gf2_gauss_jordan_kernel(const float* __restrict__ Hf, float* __restrict__ out) {
    __shared__ unsigned Hb[NROW * NWRD];   // 64 KB bit matrix
    __shared__ unsigned colsum[NCOL];      // 4 KB column popcounts
    __shared__ int      col_swap[NCOL];    // 4 KB permutation
    __shared__ int      s_max, s_piv, s_min;

    const int tid  = threadIdx.x;          // 0..511, one thread per row
    const int lane = tid & 31;
    const int wv   = tid >> 5;             // wave id, 0..15

    // ---------------- Phase 0: bit-pack H into LDS via wave32 ballot ----------
    {
        const int r0 = wv * 32;            // 32 rows per wave
        for (int r = r0; r < r0 + 32; ++r) {
            for (int wc = 0; wc < NWRD; ++wc) {
                float v = Hf[r * NCOL + wc * 32 + lane];
                unsigned mask = __builtin_amdgcn_ballot_w32(v > 0.5f);
                if (lane == 0) Hb[r * NWRD + wc] = mask;
            }
        }
    }
    if (tid == 0) s_piv = NROW;
    __syncthreads();

    // ---------------- Phase 1: right-to-left elimination ----------------------
    for (int col = NCOL - 1; col >= 0; --col) {
        if (tid == 0) s_max = -1;
        __syncthreads();

        const int wcol = col >> 5, bsh = col & 31;
        const int piv  = s_piv;
        const unsigned myw = Hb[tid * NWRD + wcol];
        if (((myw >> bsh) & 1u) && tid < piv) atomicMax(&s_max, tid);
        __syncthreads();

        const int m = s_max;
        const int found = (m >= 0);
        const int p = piv - 1;

        if (found && m != p && tid < NWRD) {   // swap rows p <-> m
            unsigned t0 = Hb[p * NWRD + tid];
            Hb[p * NWRD + tid] = Hb[m * NWRD + tid];
            Hb[m * NWRD + tid] = t0;
        }
        __syncthreads();

        if (found) {
            const unsigned rbit = (Hb[tid * NWRD + wcol] >> bsh) & 1u;
            if (rbit && tid != p) {
                #pragma unroll
                for (int w = 0; w < NWRD; ++w)      // pivot row read = LDS broadcast
                    Hb[tid * NWRD + w] ^= Hb[p * NWRD + w];
            }
        }
        __syncthreads();
        if (found && tid == 0) s_piv = piv - 1;
        __syncthreads();
    }

    // ---------------- Phase 2a: colsum = ones^T * H via WMMA ------------------
    for (int c = tid; c < NCOL; c += 512) col_swap[c] = c;
    __syncthreads();
    {
        h16 ones;
        #pragma unroll
        for (int e = 0; e < 16; ++e) ones[e] = (_Float16)1.0f;
        const int g = (lane >> 4) & 1;
        const int n = lane & 15;
        for (int blk = 0; blk < 4; ++blk) {        // 4 x 16 columns per wave
            const int colbase = (wv * 4 + blk) * 16;
            const int colj = colbase + n;
            const int wrd = colj >> 5, sh = colj & 31;
            f8 acc = {};
            for (int kb = 0; kb < 16; ++kb) {      // 16 blocks of 32 rows
                h16 b;
                #pragma unroll
                for (int e = 0; e < 16; ++e) {     // B-matrix 32x16 f16 layout
                    const int v = e >> 1, pb = e & 1;
                    const int Kb = (v < 4) ? (2 * v) : (16 + 2 * (v - 4));
                    const int K  = Kb + pb + 8 * g;
                    const int row = kb * 32 + K;
                    unsigned bitv = (Hb[row * NWRD + wrd] >> sh) & 1u;
                    b[e] = (_Float16)(float)bitv;
                }
                acc = __builtin_amdgcn_wmma_f32_16x16x32_f16(
                    false, ones, false, b, (short)0, acc, false, false);
            }
            if (lane < 16)                          // D row M=0: lanes 0..15 = N
                colsum[colbase + lane] = (unsigned)(acc[0] + 0.5f);
        }
    }
    __syncthreads();

    // ---------------- Phase 2b: unit-column identification + column swaps -----
    for (int i = 0; i < NROW; ++i) {
        if (tid == 0) s_min = 0x7fffffff;
        __syncthreads();
        for (int c = tid; c < NCOL; c += 512) {
            // column c == e_i  <=>  colsum[c]==1 && H[i][c]==1
            if (colsum[c] == 1u && ((Hb[i * NWRD + (c >> 5)] >> (c & 31)) & 1u))
                atomicMin(&s_min, c);
        }
        __syncthreads();
        const int j   = s_min;
        const int cur = (NCOL - NROW) + i;
        if (j < NCOL) {                             // found: swap columns j <-> cur
            const int wj = j >> 5,   bj = j & 31;
            const int wc = cur >> 5, bc = cur & 31;
            const unsigned vj = (Hb[tid * NWRD + wj] >> bj) & 1u;
            const unsigned vc = (Hb[tid * NWRD + wc] >> bc) & 1u;
            if (vj != vc) {
                Hb[tid * NWRD + wj] ^= (1u << bj);
                Hb[tid * NWRD + wc] ^= (1u << bc);
            }
            if (tid == 0) {
                unsigned t1 = colsum[j]; colsum[j] = colsum[cur]; colsum[cur] = t1;
                int t2 = col_swap[j];    col_swap[j] = col_swap[cur]; col_swap[cur] = t2;
            }
        }
        __syncthreads();
    }

    // ---------------- Phase 3: G_nsys = scatter([I | P^T]); src_index ---------
    for (int idx = tid; idx < NROW * NCOL; idx += 512) {
        const int r = idx >> 10, c = idx & (NCOL - 1);
        float val;
        if (c < NROW) {
            val = (r == c) ? 1.0f : 0.0f;                       // I_k
        } else {
            // H_parity^T[r][c-k] = H[c-k][r] = bit r of packed row (c-512)
            val = (float)((Hb[(c - NROW) * NWRD + (r >> 5)] >> (r & 31)) & 1u);
        }
        out[r * NCOL + col_swap[c]] = val;
    }
    out[NROW * NCOL + tid] = (float)col_swap[tid];              // src_index
}

extern "C" void kernel_launch(void* const* d_in, const int* in_sizes, int n_in,
                              void* d_out, int out_size, void* d_ws, size_t ws_size,
                              hipStream_t stream) {
    (void)in_sizes; (void)n_in; (void)out_size; (void)d_ws; (void)ws_size;
    const float* H = (const float*)d_in[0];
    float* out = (float*)d_out;
    // Single workgroup: the algorithm is sequential in columns; all state lives
    // in one WGP's LDS (72 KB of the 320 KB available).
    gf2_gauss_jordan_kernel<<<dim3(1), dim3(512), 0, stream>>>(H, out);
}
